// ProgressiveBJointBlock_5875515261423
// MI455X (gfx1250) — matrix operand order repaired and checked by reading, not compiled, not run on hardware
//
#include <hip/hip_runtime.h>
#include <hip/hip_bf16.h>

// Problem dims (match reference config)
#define Bn 4
#define Sn 2048
#define En 4096
#define Cn 512
#define Dn 256
#define WAVES 4   // waves per route workgroup

typedef unsigned short u16;
typedef __attribute__((ext_vector_type(16))) __bf16 v16bf;
typedef __attribute__((ext_vector_type(8)))  float  v8f;

union FragBF { v16bf v; uint4 q[2]; };

__device__ __forceinline__ u16 f2bf(float f) {
  unsigned int u = __float_as_uint(f);
  unsigned int r = u + 0x7FFFu + ((u >> 16) & 1u);
  return (u16)(r >> 16);
}

__device__ __forceinline__ float wave_sum(float v) {
#pragma unroll
  for (int off = 16; off > 0; off >>= 1) v += __shfl_xor(v, off, 32);
  return v;
}

// Branch-free sorted top-4 insert (v_cmp + v_cndmask chains, no exec divergence)
__device__ __forceinline__ void ins4(float v, int id,
                                     float& t0, float& t1, float& t2, float& t3,
                                     int& i0, int& i1, int& i2, int& i3) {
  bool b0 = v > t0, b1 = v > t1, b2 = v > t2, b3 = v > t3;
  float nt0 = b0 ? v : t0;              int ni0 = b0 ? id : i0;
  float nt1 = b0 ? t0 : (b1 ? v : t1);  int ni1 = b0 ? i0 : (b1 ? id : i1);
  float nt2 = b1 ? t1 : (b2 ? v : t2);  int ni2 = b1 ? i1 : (b2 ? id : i2);
  float nt3 = b2 ? t2 : (b3 ? v : t3);  int ni3 = b2 ? i2 : (b3 ? id : i3);
  t0 = nt0; t1 = nt1; t2 = nt2; t3 = nt3;
  i0 = ni0; i1 = ni1; i2 = ni2; i3 = ni3;
}

// ---------------------------------------------------------------------------
// Stage 1: window-sparse propagation. One wave per (b, s) row.
// ---------------------------------------------------------------------------
__global__ __launch_bounds__(32) void window_prop_kernel(
    const float* __restrict__ x, const float* __restrict__ w,
    const float* __restrict__ wB2s,
    float* __restrict__ s1F, u16* __restrict__ s1Bf, u16* __restrict__ s1WBf) {
  const int s = blockIdx.x, b = blockIdx.y, L = threadIdx.x;
  const int d = L * 8;
  const float* xb = x + (size_t)b * Sn * Dn;
  float xr[8], wv[8];
#pragma unroll
  for (int j = 0; j < 8; ++j) { xr[j] = xb[(size_t)s * Dn + d + j]; wv[j] = w[d + j]; }
  float nb[9][8], sc[9];
#pragma unroll
  for (int o = 0; o < 9; ++o) {
    int sn = s + o - 4;
    bool ok = (sn >= 0) && (sn < Sn);
    float part = 0.f;
#pragma unroll
    for (int j = 0; j < 8; ++j) {
      float nv = ok ? xb[(size_t)sn * Dn + d + j] : 0.f;
      nb[o][j] = nv;
      part += xr[j] * wv[j] * nv;
    }
    part = wave_sum(part);
    sc[o] = ok ? part : -1e9f;
  }
  float mx = sc[0];
#pragma unroll
  for (int o = 1; o < 9; ++o) mx = fmaxf(mx, sc[o]);
  float es[9], ssum = 0.f;
#pragma unroll
  for (int o = 0; o < 9; ++o) { es[o] = __expf(sc[o] - mx); ssum += es[o]; }
  float inv = 1.f / ssum;
  size_t ro = ((size_t)b * Sn + s) * Dn + d;
#pragma unroll
  for (int j = 0; j < 8; ++j) {
    float acc = 0.f;
#pragma unroll
    for (int o = 0; o < 9; ++o) acc += es[o] * nb[o][j];
    float y = xr[j] + 0.25f * inv * acc;
    s1F[ro + j] = y;
    s1Bf[ro + j] = f2bf(y);
    s1WBf[ro + j] = f2bf(y * wB2s[d + j]);
  }
}

// bf16(in * w[d])  (for pos-enc A operands)
__global__ void scale_cvt_kernel(const float* __restrict__ in,
                                 const float* __restrict__ w,
                                 u16* __restrict__ out, int total) {
  int i = blockIdx.x * blockDim.x + threadIdx.x;
  if (i < total) out[i] = f2bf(in[i] * w[i & (Dn - 1)]);
}

// ---------------------------------------------------------------------------
// Core: fused  logits-GEMM (bf16 WMMA) -> running row top-4 -> softmax ->
// f32 gather/combine.  4 waves/WG share one 16-row dst tile and round-robin
// the N loop; B fragments are double-buffered so each tile has a single
// loadcnt wait and the 8 WMMAs issue back-to-back while the next tile's
// loads are in flight.  Logits never touch global memory.
// ---------------------------------------------------------------------------
__global__ __launch_bounds__(32 * WAVES) void route_topk_wmma_kernel(
    const u16* __restrict__ Abf, long aBatchStride,   // 0 => A shared across batch
    const u16* __restrict__ srcBf, const float* __restrict__ srcF, int N,
    int Md,
    const float* __restrict__ base, int baseShared, float scale,
    float* __restrict__ outF,
    u16* __restrict__ outBf,
    const float* __restrict__ wOut, u16* __restrict__ outWbf) {
  const int b = blockIdx.y;
  const int m0 = blockIdx.x * 16;
  const int tid = threadIdx.x;
  const int wv = tid >> 5;
  const int L = tid & 31;
  const int lane16 = L & 15;
  const int half = L >> 4;

  __shared__ float tile[WAVES][16 * 16];
  __shared__ float mVal[WAVES][16][4];
  __shared__ int   mIdx[WAVES][16][4];
  __shared__ float aSh[16][4];
  __shared__ int   iSh[16][4];

  // A fragments (16 rows x 256 K): lanes 0-15 K={0..7,16..23}, lanes 16-31 +8
  FragBF afr[8];
  {
    const u16* arow = Abf + (size_t)b * aBatchStride + (size_t)(m0 + lane16) * Dn + half * 8;
#pragma unroll
    for (int kk = 0; kk < 8; ++kk) {
      afr[kk].q[0] = *(const uint4*)(arow + kk * 32);
      afr[kk].q[1] = *(const uint4*)(arow + kk * 32 + 16);
    }
  }

  float tv0 = -3e38f, tv1 = -3e38f, tv2 = -3e38f, tv3 = -3e38f;
  int ti0 = 0, ti1 = 0, ti2 = 0, ti3 = 0;

  // B fragment lane base: row = n0 + lane16, K = {0..7,16..23} + 8*half
  const u16* sbl = srcBf + (size_t)b * N * Dn + (size_t)lane16 * Dn + half * 8;

  auto loadB = [&](FragBF* dst, int n0) {
    const u16* p = sbl + (size_t)n0 * Dn;
#pragma unroll
    for (int kk = 0; kk < 8; ++kk) {
      dst[kk].q[0] = *(const uint4*)(p + kk * 32);
      dst[kk].q[1] = *(const uint4*)(p + kk * 32 + 16);
    }
  };

  auto processTile = [&](FragBF* bf, int n0) {
    v8f c = {};
#pragma unroll
    for (int kk = 0; kk < 8; ++kk)
      c = __builtin_amdgcn_wmma_f32_16x16x32_bf16(false, afr[kk].v, false, bf[kk].v,
                                                  (short)0, c, false, false);
    float* tl = &tile[wv][0];
    asm volatile("" ::: "memory");
#pragma unroll
    for (int r = 0; r < 8; ++r) tl[(r + 8 * half) * 16 + lane16] = c[r];
    asm volatile("" ::: "memory");
    // both halves scan in parallel: half h takes columns h*8 .. h*8+7 of its row
    const float* rp = &tile[wv][lane16 * 16 + half * 8];
    float4 q0 = *(const float4*)rp;
    float4 q1 = *(const float4*)(rp + 4);
    int idb = n0 + half * 8;
    ins4(q0.x, idb + 0, tv0, tv1, tv2, tv3, ti0, ti1, ti2, ti3);
    ins4(q0.y, idb + 1, tv0, tv1, tv2, tv3, ti0, ti1, ti2, ti3);
    ins4(q0.z, idb + 2, tv0, tv1, tv2, tv3, ti0, ti1, ti2, ti3);
    ins4(q0.w, idb + 3, tv0, tv1, tv2, tv3, ti0, ti1, ti2, ti3);
    ins4(q1.x, idb + 4, tv0, tv1, tv2, tv3, ti0, ti1, ti2, ti3);
    ins4(q1.y, idb + 5, tv0, tv1, tv2, tv3, ti0, ti1, ti2, ti3);
    ins4(q1.z, idb + 6, tv0, tv1, tv2, tv3, ti0, ti1, ti2, ti3);
    ins4(q1.w, idb + 7, tv0, tv1, tv2, tv3, ti0, ti1, ti2, ti3);
    asm volatile("" ::: "memory");
  };

  // Round-robin the n-tiles across the 4 waves, double-buffered.
  const int T = N / 16;
  FragBF bufA[8], bufB[8];
  loadB(bufA, wv * 16);
  for (int t = wv; t < T; t += 2 * WAVES) {
    int tn = t + WAVES;
    loadB(bufB, (tn < T ? tn : t) * 16);
    processTile(bufA, t * 16);
    if (tn < T) {
      int tn2 = tn + WAVES;
      loadB(bufA, (tn2 < T ? tn2 : tn) * 16);
      processTile(bufB, tn * 16);
    }
  }

  // Merge the two lane-half lists within each wave (via shuffles), then the
  // 4 per-wave lists through LDS.
  {
    float p0 = __shfl(tv0, L + 16, 32), p1 = __shfl(tv1, L + 16, 32);
    float p2 = __shfl(tv2, L + 16, 32), p3 = __shfl(tv3, L + 16, 32);
    int j0 = __shfl(ti0, L + 16, 32), j1 = __shfl(ti1, L + 16, 32);
    int j2 = __shfl(ti2, L + 16, 32), j3 = __shfl(ti3, L + 16, 32);
    if (L < 16) {
      ins4(p0, j0, tv0, tv1, tv2, tv3, ti0, ti1, ti2, ti3);
      ins4(p1, j1, tv0, tv1, tv2, tv3, ti0, ti1, ti2, ti3);
      ins4(p2, j2, tv0, tv1, tv2, tv3, ti0, ti1, ti2, ti3);
      ins4(p3, j3, tv0, tv1, tv2, tv3, ti0, ti1, ti2, ti3);
      mVal[wv][L][0] = tv0; mVal[wv][L][1] = tv1; mVal[wv][L][2] = tv2; mVal[wv][L][3] = tv3;
      mIdx[wv][L][0] = ti0; mIdx[wv][L][1] = ti1; mIdx[wv][L][2] = ti2; mIdx[wv][L][3] = ti3;
    }
  }
  __syncthreads();

  if (wv == 0 && L < 16) {
    float g0 = -3e38f, g1 = -3e38f, g2 = -3e38f, g3 = -3e38f;
    int h0 = 0, h1 = 0, h2 = 0, h3 = 0;
#pragma unroll
    for (int q = 0; q < WAVES; ++q)
#pragma unroll
      for (int k = 0; k < 4; ++k)
        ins4(mVal[q][L][k], mIdx[q][L][k], g0, g1, g2, g3, h0, h1, h2, h3);
    // softmax over the selected top-4 edges
    float x1 = __expf(g1 - g0), x2 = __expf(g2 - g0), x3 = __expf(g3 - g0);
    float inv = 1.f / (1.f + x1 + x2 + x3);
    aSh[L][0] = inv; aSh[L][1] = x1 * inv; aSh[L][2] = x2 * inv; aSh[L][3] = x3 * inv;
    iSh[L][0] = h0; iSh[L][1] = h1; iSh[L][2] = h2; iSh[L][3] = h3;
  }
  __syncthreads();

  // f32 gather/combine: each wave takes 4 rows, 8 features/lane
  const float* sf = srcF + (size_t)b * N * Dn;
  const int d = L * 8;
  for (int r = wv * 4; r < wv * 4 + 4; ++r) {
    float acc[8] = {0, 0, 0, 0, 0, 0, 0, 0};
#pragma unroll
    for (int k = 0; k < 4; ++k) {
      float a = aSh[r][k];
      const float* sp = sf + (size_t)iSh[r][k] * Dn + d;
      float4 p0 = *(const float4*)sp;
      float4 p1 = *(const float4*)(sp + 4);
      acc[0] += a * p0.x; acc[1] += a * p0.y; acc[2] += a * p0.z; acc[3] += a * p0.w;
      acc[4] += a * p1.x; acc[5] += a * p1.y; acc[6] += a * p1.z; acc[7] += a * p1.w;
    }
    size_t orow = ((size_t)b * Md + m0 + r) * Dn + d;
    size_t bidx = baseShared ? ((size_t)(m0 + r) * Dn + d) : orow;
#pragma unroll
    for (int j = 0; j < 8; ++j) {
      float v = acc[j];
      if (base) v = base[bidx + j] + scale * v;
      outF[orow + j] = v;
      if (outBf)  outBf[orow + j]  = f2bf(v);
      if (outWbf) outWbf[orow + j] = f2bf(v * wOut[d + j]);
    }
  }
}

// ---------------------------------------------------------------------------
// Fused  t = base + alpha*r ; LN(t)  with optional bf16 / w-scaled bf16 copies.
// ---------------------------------------------------------------------------
__global__ __launch_bounds__(32) void ln_fuse_kernel(
    const float* __restrict__ base, int baseShared, int Md,
    const float* __restrict__ rsrc, float alpha,
    const float* __restrict__ gamma, const float* __restrict__ beta,
    float* __restrict__ outF,
    u16* __restrict__ outBf,
    const float* __restrict__ wvec, u16* __restrict__ outWbf) {
  int row = blockIdx.x;  // 0 .. B*Md-1
  int m = row % Md;
  int L = threadIdx.x, d = L * 8;
  size_t ro = (size_t)row * Dn + d;
  size_t bo = baseShared ? ((size_t)m * Dn + d) : ro;
  float t[8];
#pragma unroll
  for (int j = 0; j < 8; ++j) t[j] = base[bo + j] + alpha * rsrc[ro + j];
  float s = 0.f;
#pragma unroll
  for (int j = 0; j < 8; ++j) s += t[j];
  s = wave_sum(s);
  float mean = s * (1.0f / Dn);
  float vs = 0.f;
#pragma unroll
  for (int j = 0; j < 8; ++j) { float c = t[j] - mean; vs += c * c; }
  vs = wave_sum(vs);
  float inv = rsqrtf(vs * (1.0f / Dn) + 1e-5f);
#pragma unroll
  for (int j = 0; j < 8; ++j) {
    float y = (t[j] - mean) * inv * gamma[d + j] + beta[d + j];
    outF[ro + j] = y;
    if (outBf)  outBf[ro + j]  = f2bf(y);
    if (outWbf) outWbf[ro + j] = f2bf(y * wvec[d + j]);
  }
}

extern "C" void kernel_launch(void* const* d_in, const int* in_sizes, int n_in,
                              void* d_out, int out_size, void* d_ws, size_t ws_size,
                              hipStream_t stream) {
  (void)in_sizes; (void)n_in; (void)out_size; (void)ws_size;
  const float* x      = (const float*)d_in[0];
  const float* w_s    = (const float*)d_in[1];
  const float* pe_exp = (const float*)d_in[2];
  const float* w_er   = (const float*)d_in[3];
  const float* w_ep   = (const float*)d_in[4];
  const float* g_e    = (const float*)d_in[5];
  const float* b_e    = (const float*)d_in[6];
  const float* w_b2s  = (const float*)d_in[7];
  const float* pe_c   = (const float*)d_in[8];
  const float* w_cr   = (const float*)d_in[9];
  const float* w_cp   = (const float*)d_in[10];
  const float* g_c    = (const float*)d_in[11];
  const float* b_c    = (const float*)d_in[12];
  const float* w_c2s  = (const float*)d_in[13];
  const float* g_s    = (const float*)d_in[14];
  const float* b_s    = (const float*)d_in[15];
  float* out = (float*)d_out;

  char* wsp = (char*)d_ws;
  size_t off = 0;
  auto alloc = [&](size_t bytes) -> void* {
    void* q = wsp + off;
    off = (off + bytes + 255) & ~(size_t)255;
    return q;
  };
  const size_t NS = (size_t)Bn * Sn * Dn, NE = (size_t)Bn * En * Dn, NC = (size_t)Bn * Cn * Dn;
  float* s1F   = (float*)alloc(NS * 4);
  u16*   s1Bf  = (u16*)alloc(NS * 2);
  u16*   s1WBf = (u16*)alloc(NS * 2);              // s1 * w_b_to_s   (A for b->s)
  u16*   peEW  = (u16*)alloc((size_t)En * Dn * 2); // pe_exp * w_expand_route
  u16*   peCW  = (u16*)alloc((size_t)Cn * Dn * 2); // pe_comp * w_compress_route
  float* r1F   = (float*)alloc(NE * 4);
  float* e1F   = (float*)alloc(NE * 4);
  u16*   e1Bf  = (u16*)alloc(NE * 2);
  u16*   e1WBf = (u16*)alloc(NE * 2);              // e1 * w_exp_pair
  float* e2F   = (float*)alloc(NE * 4);
  u16*   e2Bf  = (u16*)alloc(NE * 2);
  float* s2F   = (float*)alloc(NS * 4);
  u16*   s2Bf  = (u16*)alloc(NS * 2);
  u16*   s2WBf = (u16*)alloc(NS * 2);              // s2 * w_c_to_s
  float* rcF   = (float*)alloc(NC * 4);
  float* c1F   = (float*)alloc(NC * 4);
  u16*   c1Bf  = (u16*)alloc(NC * 2);
  u16*   c1WBf = (u16*)alloc(NC * 2);              // c1 * w_comp_pair
  float* c2F   = (float*)alloc(NC * 4);
  u16*   c2Bf  = (u16*)alloc(NC * 2);
  float* s3F   = (float*)alloc(NS * 4);

  // 1) s1 = window_prop(x) (+ bf16 side copies)
  window_prop_kernel<<<dim3(Sn, Bn), 32, 0, stream>>>(x, w_s, w_b2s, s1F, s1Bf, s1WBf);
  // pos-enc A operands
  scale_cvt_kernel<<<(En * Dn + 255) / 256, 256, 0, stream>>>(pe_exp, w_er, peEW, En * Dn);
  scale_cvt_kernel<<<(Cn * Dn + 255) / 256, 256, 0, stream>>>(pe_c, w_cr, peCW, Cn * Dn);
  // 2) r1 = route(s1 -> expanded), raw combine
  route_topk_wmma_kernel<<<dim3(En / 16, Bn), 32 * WAVES, 0, stream>>>(
      peEW, 0, s1Bf, s1F, Sn, En, nullptr, 0, 0.f, r1F, nullptr, nullptr, nullptr);
  // 3) e1 = LN(pe_exp + 0.5*r1)
  ln_fuse_kernel<<<Bn * En, 32, 0, stream>>>(pe_exp, 1, En, r1F, 0.5f, g_e, b_e,
                                             e1F, e1Bf, w_ep, e1WBf);
  // 4) e2 = e1 + 0.2 * topk_prop(e1)      (E x E logits, fused top-4)
  route_topk_wmma_kernel<<<dim3(En / 16, Bn), 32 * WAVES, 0, stream>>>(
      e1WBf, (long)En * Dn, e1Bf, e1F, En, En, e1F, 0, 0.2f, e2F, e2Bf, nullptr, nullptr);
  // 5) s2 = s1 + 0.5*0.15 * route(e2 -> s1)
  route_topk_wmma_kernel<<<dim3(Sn / 16, Bn), 32 * WAVES, 0, stream>>>(
      s1WBf, (long)Sn * Dn, e2Bf, e2F, En, Sn, s1F, 0, 0.075f, s2F, s2Bf, w_c2s, s2WBf);
  // 6) rc = route(s2 -> compressed), raw combine
  route_topk_wmma_kernel<<<dim3(Cn / 16, Bn), 32 * WAVES, 0, stream>>>(
      peCW, 0, s2Bf, s2F, Sn, Cn, nullptr, 0, 0.f, rcF, nullptr, nullptr, nullptr);
  // 7) c1 = LN(pe_comp + 0.15*rc)
  ln_fuse_kernel<<<Bn * Cn, 32, 0, stream>>>(pe_c, 1, Cn, rcF, 0.15f, g_c, b_c,
                                             c1F, c1Bf, w_cp, c1WBf);
  // 8) c2 = c1 + 0.2 * topk_prop(c1)
  route_topk_wmma_kernel<<<dim3(Cn / 16, Bn), 32 * WAVES, 0, stream>>>(
      c1WBf, (long)Cn * Dn, c1Bf, c1F, Cn, Cn, c1F, 0, 0.2f, c2F, c2Bf, nullptr, nullptr);
  // 9) s3 = s2 + 0.5*0.15 * route(c2 -> s2)
  route_topk_wmma_kernel<<<dim3(Sn / 16, Bn), 32 * WAVES, 0, stream>>>(
      s2WBf, (long)Sn * Dn, c2Bf, c2F, Cn, Sn, s2F, 0, 0.075f, s3F, nullptr, nullptr, nullptr);
  // 10) out = LN(s3)
  ln_fuse_kernel<<<Bn * Sn, 32, 0, stream>>>(s3F, 0, Sn, s3F, 0.f, g_s, b_s,
                                             out, nullptr, nullptr, nullptr);
}
